// VisionTransformerBlock_26113401160326
// MI455X (gfx1250) — matrix operand order repaired and checked by reading, not compiled, not run on hardware
//
#include <hip/hip_runtime.h>
#include <math.h>

typedef __attribute__((ext_vector_type(8)))  float    v8f;
typedef __attribute__((ext_vector_type(16))) _Float16 v16h;
typedef __attribute__((ext_vector_type(2)))  _Float16 h2;
typedef __attribute__((ext_vector_type(4)))  _Float16 h4;

#define Bdim 16
#define Pdim 2048
#define Edim 768
#define Hdim 256
#define QKVW (3 * Hdim)        // 768
#define NPB (Pdim * Edim)      // 1572864 elems per batch

// GEMM tiling: 256 threads = 8 waves; block tile 128(M) x 64(N), K-step 32.
// Each wave: 2x2 grid of 16x16 WMMA subtiles via v_wmma_f32_16x16x32_f16.
#define TM 128
#define TN 64
#define TK 32
#define LDA_H 36               // halves per A-tile row (pad: 72B rows, 8B aligned)

// ---- fragment loaders (ISA 7.12.2 16-bit layouts) -------------------------
// A (16x32, MK layout in LDS): VGPR r(0..3): K = 8*hi + 2r,+1 ;
//                              VGPR 4+s   : K = 16 + 8*hi + 2s,+1
__device__ __forceinline__ v16h load_a_mk(const _Float16* la, int m, int hi) {
  const h2* w = (const h2*)la;           // word = 2 consecutive K halves
  const int base = m * (LDA_H / 2);      // 18 words per row
  v16h a;
#pragma unroll
  for (int r = 0; r < 4; ++r) {
    const h2 p = w[base + hi * 4 + r];
    a[2 * r] = p[0];
    a[2 * r + 1] = p[1];
  }
#pragma unroll
  for (int s = 0; s < 4; ++s) {
    const h2 p = w[base + 8 + hi * 4 + s];
    a[8 + 2 * s] = p[0];
    a[8 + 2 * s + 1] = p[1];
  }
  return a;
}

// A in paired-KM layout: halves (k,k+1) of column m at word (k>>1)*TM + m
__device__ __forceinline__ v16h load_a_km(const _Float16* la, int m, int hi) {
  const h2* w = (const h2*)la;
  v16h a;
#pragma unroll
  for (int r = 0; r < 4; ++r) {
    const h2 p = w[(hi * 4 + r) * TM + m];
    a[2 * r] = p[0];
    a[2 * r + 1] = p[1];
  }
#pragma unroll
  for (int s = 0; s < 4; ++s) {
    const h2 p = w[(8 + hi * 4 + s) * TM + m];
    a[8 + 2 * s] = p[0];
    a[8 + 2 * s + 1] = p[1];
  }
  return a;
}

// B (32x16): VGPR j<4: K = 8*hi + 2j,+1 ; j>=4: K = 16 + 8*hi + 2(j-4),+1 at column n. Paired layout:
// halves (k,k+1) of column n at word (k>>1)*TN + n.
__device__ __forceinline__ v16h load_b(const _Float16* lb, int n, int hi) {
  const h2* w = (const h2*)lb;
  v16h b;
#pragma unroll
  for (int j = 0; j < 8; ++j) {
    const h2 p = w[(hi * 4 + j + (j & 4)) * TN + n];
    b[2 * j] = p[0];
    b[2 * j + 1] = p[1];
  }
  return b;
}

template <int AKM>
__device__ __forceinline__ void wave_mac(const _Float16* __restrict__ la,
                                         const _Float16* __restrict__ lb,
                                         int mbase, int nbase, int lo, int hi,
                                         v8f acc[2][2]) {
  v16h a[2], b[2];
#pragma unroll
  for (int sm = 0; sm < 2; ++sm) {
    const int m = mbase + sm * 16 + lo;
    a[sm] = AKM ? load_a_km(la, m, hi) : load_a_mk(la, m, hi);
  }
#pragma unroll
  for (int sn = 0; sn < 2; ++sn) b[sn] = load_b(lb, nbase + sn * 16 + lo, hi);
#pragma unroll
  for (int sm = 0; sm < 2; ++sm)
#pragma unroll
    for (int sn = 0; sn < 2; ++sn)
      acc[sm][sn] = __builtin_amdgcn_wmma_f32_16x16x32_f16(
          false, a[sm], false, b[sn], (short)0, acc[sm][sn], false, false);
}

__device__ __forceinline__ void store_tile(float* __restrict__ C, int ldc,
                                           int row0, int col0, int lo, int hi,
                                           const v8f& acc) {
  const int col = col0 + lo;
  const int rbase = row0 + (hi ? 8 : 0);
#pragma unroll
  for (int r = 0; r < 8; ++r) C[(rbase + r) * ldc + col] = acc[r];
}

// paired B-tile store helper: element (k,n) -> lb[((k>>1)*TN + n)*2 + (k&1)]
__device__ __forceinline__ void stb(_Float16* lb, int k, int n, float v) {
  lb[(((k >> 1) * TN) + n) * 2 + (k & 1)] = (_Float16)v;
}

// ---------------- LayerNorm statistics (two-stage, deterministic) ----------
#define RCH 64                       // chunks per batch
#define RCHUNK (NPB / RCH)           // 24576 floats per block

__global__ __launch_bounds__(256) void k_reduce1(const float* __restrict__ x,
                                                 float* __restrict__ part) {
  __shared__ float ss[256], sq[256];
  const int tid = threadIdx.x;
  const float4* x4 = (const float4*)x + (size_t)blockIdx.x * (RCHUNK / 4);
  float s = 0.f, q = 0.f;
  for (int i = tid; i < RCHUNK / 4; i += 256) {
    const float4 v = x4[i];
    s += v.x + v.y + v.z + v.w;
    q += v.x * v.x + v.y * v.y + v.z * v.z + v.w * v.w;
  }
  ss[tid] = s;
  sq[tid] = q;
  __syncthreads();
  for (int st = 128; st > 0; st >>= 1) {
    if (tid < st) {
      ss[tid] += ss[tid + st];
      sq[tid] += sq[tid + st];
    }
    __syncthreads();
  }
  if (tid == 0) {
    part[blockIdx.x * 2 + 0] = ss[0];
    part[blockIdx.x * 2 + 1] = sq[0];
  }
}

__global__ __launch_bounds__(64) void k_reduce2(const float* __restrict__ part,
                                                float* __restrict__ stats) {
  __shared__ float ss[64], sq[64];
  const int tid = threadIdx.x;
  const int b = blockIdx.x;
  ss[tid] = part[(b * RCH + tid) * 2 + 0];
  sq[tid] = part[(b * RCH + tid) * 2 + 1];
  __syncthreads();
  for (int st = 32; st > 0; st >>= 1) {
    if (tid < st) {
      ss[tid] += ss[tid + st];
      sq[tid] += sq[tid + st];
    }
    __syncthreads();
  }
  if (tid == 0) {
    const float n = (float)NPB;
    const float mean = ss[0] / n;
    const float var = sq[0] / n - mean * mean;
    stats[b * 2 + 0] = mean;
    stats[b * 2 + 1] = rsqrtf(var + 1e-5f);
  }
}

// ---------------- QKV GEMM with fused LayerNorm + f32->f16 on A ------------
__global__ __launch_bounds__(256) void k_qkv_gemm(
    const float* __restrict__ x, const float* __restrict__ nw,
    const float* __restrict__ nb, const float* __restrict__ qkv_w,
    const float* __restrict__ stats, float* __restrict__ qkv) {
  __shared__ _Float16 la[TM * LDA_H];
  __shared__ _Float16 lb[(TK / 2) * TN * 2];

  const int tid = threadIdx.x;
  const int lane = tid & 31, w = tid >> 5;
  const int lo = lane & 15, hi = lane >> 4;
  const int wm = (w & 3) * 32, wn = (w >> 2) * 32;

  const int mtile = blockIdx.y * TM;   // global row, all within one batch
  const int ntile = blockIdx.x * TN;
  const int b = mtile / Pdim;
  const float mean = stats[b * 2 + 0];
  const float invstd = stats[b * 2 + 1];

  v8f acc[2][2] = {};
  for (int k0 = 0; k0 < Edim; k0 += TK) {
#pragma unroll
    for (int i = 0; i < 4; ++i) {        // A tile: 128x32 = 1024 float4
      const int f = tid + i * 256;
      const int row = f >> 3;
      const int colv = (f & 7) << 2;
      const int rg = mtile + row;
      const int p = rg - b * Pdim;
      const float4 xv = *(const float4*)(x + rg * Edim + k0 + colv);
      const float4 wv = *(const float4*)(nw + p * Edim + k0 + colv);
      const float4 bv = *(const float4*)(nb + p * Edim + k0 + colv);
      h4 hv;
      hv[0] = (_Float16)((xv.x - mean) * invstd * wv.x + bv.x);
      hv[1] = (_Float16)((xv.y - mean) * invstd * wv.y + bv.y);
      hv[2] = (_Float16)((xv.z - mean) * invstd * wv.z + bv.z);
      hv[3] = (_Float16)((xv.w - mean) * invstd * wv.w + bv.w);
      *(h4*)(la + row * LDA_H + colv) = hv;
    }
#pragma unroll
    for (int i = 0; i < 2; ++i) {        // B tile: 32x64 = 512 float4
      const int f = tid + i * 256;
      const int krow = f >> 4;
      const int colv = (f & 15) << 2;
      const float4 v = *(const float4*)(qkv_w + (k0 + krow) * QKVW + ntile + colv);
      stb(lb, krow, colv + 0, v.x);
      stb(lb, krow, colv + 1, v.y);
      stb(lb, krow, colv + 2, v.z);
      stb(lb, krow, colv + 3, v.w);
    }
    __syncthreads();
    wave_mac<0>(la, lb, wm, wn, lo, hi, acc);
    __syncthreads();
  }
#pragma unroll
  for (int sm = 0; sm < 2; ++sm)
#pragma unroll
    for (int sn = 0; sn < 2; ++sn)
      store_tile(qkv, QKVW, mtile + wm + sm * 16, ntile + wn + sn * 16, lo, hi,
                 acc[sm][sn]);
}

// ---------------- kv[b] = k[b]^T @ v[b]  (256x256, K = P = 2048) -----------
__global__ __launch_bounds__(256) void k_kv_gemm(const float* __restrict__ qkv,
                                                 float* __restrict__ kv) {
  __shared__ _Float16 la[(TK / 2) * TM * 2];   // paired KM
  __shared__ _Float16 lb[(TK / 2) * TN * 2];

  const int tid = threadIdx.x;
  const int lane = tid & 31, w = tid >> 5;
  const int lo = lane & 15, hi = lane >> 4;
  const int wm = (w & 3) * 32, wn = (w >> 2) * 32;

  const int b = blockIdx.z;
  const int itile = blockIdx.y * TM;
  const int jtile = blockIdx.x * TN;
  const float* base = qkv + b * (Pdim * QKVW);

  v8f acc[2][2] = {};
  for (int p0 = 0; p0 < Pdim; p0 += TK) {
#pragma unroll
    for (int i = 0; i < 4; ++i) {        // A tile: 32(p) x 128(i) float4s
      const int f = tid + i * 256;
      const int p = f >> 5;
      const int iv = (f & 31) << 2;
      const float4 v = *(const float4*)(base + (p0 + p) * QKVW + Hdim + itile + iv);
      _Float16* d = la + ((p >> 1) * TM) * 2 + (p & 1);
      d[(iv + 0) * 2] = (_Float16)v.x;
      d[(iv + 1) * 2] = (_Float16)v.y;
      d[(iv + 2) * 2] = (_Float16)v.z;
      d[(iv + 3) * 2] = (_Float16)v.w;
    }
#pragma unroll
    for (int i = 0; i < 2; ++i) {        // B tile: 32(p) x 64(j)
      const int f = tid + i * 256;
      const int p = f >> 4;
      const int jv = (f & 15) << 2;
      const float4 v = *(const float4*)(base + (p0 + p) * QKVW + 2 * Hdim + jtile + jv);
      stb(lb, p, jv + 0, v.x);
      stb(lb, p, jv + 1, v.y);
      stb(lb, p, jv + 2, v.z);
      stb(lb, p, jv + 3, v.w);
    }
    __syncthreads();
    wave_mac<1>(la, lb, wm, wn, lo, hi, acc);
    __syncthreads();
  }
  float* cb = kv + b * Hdim * Hdim;
#pragma unroll
  for (int sm = 0; sm < 2; ++sm)
#pragma unroll
    for (int sn = 0; sn < 2; ++sn)
      store_tile(cb, Hdim, itile + wm + sm * 16, jtile + wn + sn * 16, lo, hi,
                 acc[sm][sn]);
}

// ---------------- o[b] = q[b] @ kv[b]  (M=2048/batch, K=N=256) -------------
__global__ __launch_bounds__(256) void k_o_gemm(const float* __restrict__ qkv,
                                                const float* __restrict__ kv,
                                                float* __restrict__ o) {
  __shared__ _Float16 la[TM * LDA_H];
  __shared__ _Float16 lb[(TK / 2) * TN * 2];

  const int tid = threadIdx.x;
  const int lane = tid & 31, w = tid >> 5;
  const int lo = lane & 15, hi = lane >> 4;
  const int wm = (w & 3) * 32, wn = (w >> 2) * 32;

  const int mtile = blockIdx.y * TM;
  const int ntile = blockIdx.x * TN;
  const int b = mtile / Pdim;
  const float* kvb = kv + b * Hdim * Hdim;

  v8f acc[2][2] = {};
  for (int k0 = 0; k0 < Hdim; k0 += TK) {
#pragma unroll
    for (int i = 0; i < 4; ++i) {
      const int f = tid + i * 256;
      const int row = f >> 3;
      const int colv = (f & 7) << 2;
      const float4 v = *(const float4*)(qkv + (mtile + row) * QKVW + k0 + colv);
      h4 hv;
      hv[0] = (_Float16)v.x;
      hv[1] = (_Float16)v.y;
      hv[2] = (_Float16)v.z;
      hv[3] = (_Float16)v.w;
      *(h4*)(la + row * LDA_H + colv) = hv;
    }
#pragma unroll
    for (int i = 0; i < 2; ++i) {
      const int f = tid + i * 256;
      const int krow = f >> 4;
      const int colv = (f & 15) << 2;
      const float4 v = *(const float4*)(kvb + (k0 + krow) * Hdim + ntile + colv);
      stb(lb, krow, colv + 0, v.x);
      stb(lb, krow, colv + 1, v.y);
      stb(lb, krow, colv + 2, v.z);
      stb(lb, krow, colv + 3, v.w);
    }
    __syncthreads();
    wave_mac<0>(la, lb, wm, wn, lo, hi, acc);
    __syncthreads();
  }
#pragma unroll
  for (int sm = 0; sm < 2; ++sm)
#pragma unroll
    for (int sn = 0; sn < 2; ++sn)
      store_tile(o, Hdim, mtile + wm + sm * 16, ntile + wn + sn * 16, lo, hi,
                 acc[sm][sn]);
}

// ---------------- out = relu(o @ lin_w^T + lin_b) --------------------------
__global__ __launch_bounds__(256) void k_out_gemm(
    const float* __restrict__ o, const float* __restrict__ lin_w,
    const float* __restrict__ lin_b, float* __restrict__ out) {
  __shared__ _Float16 la[TM * LDA_H];
  __shared__ _Float16 lb[(TK / 2) * TN * 2];

  const int tid = threadIdx.x;
  const int lane = tid & 31, w = tid >> 5;
  const int lo = lane & 15, hi = lane >> 4;
  const int wm = (w & 3) * 32, wn = (w >> 2) * 32;

  const int mtile = blockIdx.y * TM;
  const int ntile = blockIdx.x * TN;   // output feature e

  v8f acc[2][2] = {};
  for (int k0 = 0; k0 < Hdim; k0 += TK) {
#pragma unroll
    for (int i = 0; i < 4; ++i) {
      const int f = tid + i * 256;
      const int row = f >> 3;
      const int colv = (f & 7) << 2;
      const float4 v = *(const float4*)(o + (mtile + row) * Hdim + k0 + colv);
      h4 hv;
      hv[0] = (_Float16)v.x;
      hv[1] = (_Float16)v.y;
      hv[2] = (_Float16)v.z;
      hv[3] = (_Float16)v.w;
      *(h4*)(la + row * LDA_H + colv) = hv;
    }
#pragma unroll
    for (int i = 0; i < 2; ++i) {
      // B[k][n] = lin_w[ntile+n][k0+k]; float4 contiguous along k
      const int f = tid + i * 256;
      const int c = f & 63;
      const int r0 = (f >> 6) << 2;
      const float4 v = *(const float4*)(lin_w + (ntile + c) * Hdim + k0 + r0);
      h2* wds = (h2*)lb;
      h2 p01, p23;
      p01[0] = (_Float16)v.x;
      p01[1] = (_Float16)v.y;
      p23[0] = (_Float16)v.z;
      p23[1] = (_Float16)v.w;
      wds[(r0 >> 1) * TN + c] = p01;           // halves (k=r0, r0+1)
      wds[((r0 >> 1) + 1) * TN + c] = p23;     // halves (k=r0+2, r0+3)
    }
    __syncthreads();
    wave_mac<0>(la, lb, wm, wn, lo, hi, acc);
    __syncthreads();
  }
#pragma unroll
  for (int sm = 0; sm < 2; ++sm)
#pragma unroll
    for (int sn = 0; sn < 2; ++sn) {
      const int col = ntile + wn + sn * 16 + lo;
      const float bias = lin_b[col];
      const int rbase = mtile + wm + sm * 16 + (hi ? 8 : 0);
#pragma unroll
      for (int r = 0; r < 8; ++r)
        out[(rbase + r) * Edim + col] = fmaxf(acc[sm][sn][r] + bias, 0.f);
    }
}

extern "C" void kernel_launch(void* const* d_in, const int* in_sizes, int n_in,
                              void* d_out, int out_size, void* d_ws,
                              size_t ws_size, hipStream_t stream) {
  const float* x = (const float*)d_in[0];
  const float* norm_w = (const float*)d_in[1];
  const float* norm_b = (const float*)d_in[2];
  const float* qkv_w = (const float*)d_in[3];
  const float* lin_w = (const float*)d_in[4];
  const float* lin_b = (const float*)d_in[5];
  float* out = (float*)d_out;

  float* ws = (float*)d_ws;
  float* part = ws;                                   // B*RCH*2 = 2048
  float* stats = part + Bdim * RCH * 2;               // B*2 = 32
  float* qkv = stats + Bdim * 2;                      // B*P*3H
  float* kv = qkv + (size_t)Bdim * Pdim * QKVW;       // B*H*H
  float* o = kv + (size_t)Bdim * Hdim * Hdim;         // B*P*H

  k_reduce1<<<dim3(Bdim * RCH), 256, 0, stream>>>(x, part);
  k_reduce2<<<dim3(Bdim), 64, 0, stream>>>(part, stats);
  k_qkv_gemm<<<dim3(QKVW / TN, (Bdim * Pdim) / TM), 256, 0, stream>>>(
      x, norm_w, norm_b, qkv_w, stats, qkv);
  k_kv_gemm<<<dim3(Hdim / TN, Hdim / TM, Bdim), 256, 0, stream>>>(qkv, kv);
  k_o_gemm<<<dim3(Hdim / TN, (Bdim * Pdim) / TM), 256, 0, stream>>>(qkv, kv, o);
  k_out_gemm<<<dim3(Edim / TN, (Bdim * Pdim) / TM), 256, 0, stream>>>(
      o, lin_w, lin_b, out);
}
